// MoELayer_78460462564088
// MI455X (gfx1250) — matrix operand order repaired and checked by reading, not compiled, not run on hardware
//
#include <hip/hip_runtime.h>
#include <hip/hip_bf16.h>

typedef __attribute__((ext_vector_type(16))) _Float16 v16h;
typedef __attribute__((ext_vector_type(8)))  _Float16 v8h;
typedef __attribute__((ext_vector_type(4)))  _Float16 v4h;
typedef __attribute__((ext_vector_type(8)))  float    v8f;

#define B_TOK 8192
#define DDIM  2048
#define HDIM  2048
#define NEXP  8
#define BM    32
#define BN    128
#define BK    128
#define APITCH 136                       // 128 + 8 halfs; row stride 272B (16B aligned)
#define MAXSLOTS (2*B_TOK + NEXP*BM)     // 16640 (bins padded to 32)
#define MAXTILES (MAXSLOTS/BM)           // 520

// ws layout: control arrays (4-byte elems) in first 1MB, then f16 expert weights
#define OFF_COUNTS   0
#define OFF_BASE     8
#define OFF_CURSOR   16
#define OFF_TOTAL    24
#define OFF_ROUTE_E  32
#define OFF_ROUTE_P  (OFF_ROUTE_E + 2*B_TOK)     // 16416
#define OFF_SLOT_TOK (OFF_ROUTE_P + 2*B_TOK)     // 32800
#define OFF_SLOT_W   (OFF_SLOT_TOK + MAXSLOTS)   // 49440
#define OFF_TILE_E   (OFF_SLOT_W + MAXSLOTS)     // 66080
#define OFF_W16_BYTES (1u << 20)                 // f16 weights at +1MB

// ---------------- init: zero output, init ws control arrays ----------------
__global__ void k_init(float* __restrict__ out, int* __restrict__ wsi,
                       float* __restrict__ wsf) {
    long long idx = (long long)blockIdx.x * blockDim.x + threadIdx.x;
    long long stride = (long long)gridDim.x * blockDim.x;
    long long n = (long long)B_TOK * HDIM;
    for (long long i = idx; i < n; i += stride) out[i] = 0.0f;
    for (long long i = idx; i < 32; i += stride) wsi[i] = 0;
    for (long long i = idx; i < MAXSLOTS; i += stride) {
        wsi[OFF_SLOT_TOK + i] = -1;
        wsf[OFF_SLOT_W  + i] = 0.0f;
    }
    for (long long i = idx; i < MAXTILES; i += stride) wsi[OFF_TILE_E + i] = -1;
}

// ---------------- convert expert weights f32 -> f16 (once) -----------------
__global__ __launch_bounds__(256) void k_cvt(const float* __restrict__ w,
                                             _Float16* __restrict__ w16) {
    long long idx = (long long)blockIdx.x * blockDim.x + threadIdx.x;
    long long stride = (long long)gridDim.x * blockDim.x;
    long long n4 = (long long)NEXP * HDIM * DDIM / 4;
    const float4* src = (const float4*)w;
    v4h* dst = (v4h*)w16;
    for (long long i = idx; i < n4; i += stride) {
        float4 v = src[i];
        v4h o;
        o[0] = (_Float16)v.x; o[1] = (_Float16)v.y;
        o[2] = (_Float16)v.z; o[3] = (_Float16)v.w;
        dst[i] = o;
    }
}

// ---------------- gate: logits, top-2, softmax, per-expert counts ----------
__global__ __launch_bounds__(256) void k_gate(const float* __restrict__ x,
                                              const float* __restrict__ gw,
                                              const float* __restrict__ gb,
                                              int* __restrict__ route_e,
                                              float* __restrict__ route_p,
                                              int* __restrict__ counts) {
    int wave = threadIdx.x >> 5;
    int lane = threadIdx.x & 31;
    int b = blockIdx.x * 8 + wave;                 // one wave32 per token
    const float* xr = x + (long long)b * DDIM;

    float acc[NEXP];
#pragma unroll
    for (int e = 0; e < NEXP; e++) acc[e] = 0.0f;

    for (int i = 0; i < DDIM / 32; i++) {
        int d = i * 32 + lane;
        float xv = xr[d];
#pragma unroll
        for (int e = 0; e < NEXP; e++) acc[e] += xv * gw[e * DDIM + d];
    }
#pragma unroll
    for (int e = 0; e < NEXP; e++)
        for (int off = 16; off > 0; off >>= 1)
            acc[e] += __shfl_down(acc[e], off);

    if (lane == 0) {
        float v[NEXP];
#pragma unroll
        for (int e = 0; e < NEXP; e++) v[e] = acc[e] + gb[e];
        int i1 = 0;
#pragma unroll
        for (int e = 1; e < NEXP; e++) if (v[e] > v[i1]) i1 = e;
        int i2 = (i1 == 0) ? 1 : 0;
#pragma unroll
        for (int e = 0; e < NEXP; e++) if (e != i1 && v[e] > v[i2]) i2 = e;
        float t  = __expf(v[i2] - v[i1]);          // <= 1, stable
        float p1 = 1.0f / (1.0f + t);
        float p2 = t * p1;
        route_e[2 * b] = i1;  route_e[2 * b + 1] = i2;
        route_p[2 * b] = p1;  route_p[2 * b + 1] = p2;
        atomicAdd(&counts[i1], 1);
        atomicAdd(&counts[i2], 1);
    }
}

// ---------------- scan: prefix sum over 8 experts, tiles of 32 -------------
__global__ void k_scan(int* __restrict__ wsi) {
    if (threadIdx.x == 0 && blockIdx.x == 0) {
        int acc = 0;
        for (int e = 0; e < NEXP; e++) {
            wsi[OFF_BASE + e] = acc;
            int c  = wsi[OFF_COUNTS + e];
            int nt = (c + BM - 1) / BM;            // tiles of 32 (padded)
            int t0 = acc / BM;
            for (int t = 0; t < nt; t++) wsi[OFF_TILE_E + t0 + t] = e;
            acc += nt * BM;
        }
        wsi[OFF_TOTAL] = acc / BM;
    }
}

// ---------------- scatter: compact tokens into per-expert slot lists -------
__global__ __launch_bounds__(256) void k_scatter(int* __restrict__ wsi,
                                                 float* __restrict__ wsf) {
    int b = blockIdx.x * blockDim.x + threadIdx.x;
    if (b >= B_TOK) return;
#pragma unroll
    for (int k = 0; k < 2; k++) {
        int   e = wsi[OFF_ROUTE_E + 2 * b + k];
        float p = wsf[OFF_ROUTE_P + 2 * b + k];
        int pos = atomicAdd(&wsi[OFF_CURSOR + e], 1);
        int s   = wsi[OFF_BASE + e] + pos;
        wsi[OFF_SLOT_TOK + s] = b;
        wsf[OFF_SLOT_W  + s] = p;
    }
}

// ------- expert GEMM: 32 tokens x 128 h per block, f16 WMMA, BK=128 --------
// 8 waves; wave w owns n-tile w (16 cols) and BOTH m-tiles (B frag reused).
// A (x rows) staged through LDS with f32->f16 convert; B frags loaded
// directly from pre-converted f16 weights (L2-resident), no LDS, no sync.
__global__ __launch_bounds__(256) void k_gemm(const float* __restrict__ x,
                                              const _Float16* __restrict__ w16,
                                              const float* __restrict__ eb,
                                              const int*   __restrict__ slot_tok,
                                              const float* __restrict__ slot_w,
                                              const int*   __restrict__ tile_e,
                                              float* __restrict__ out) {
    __shared__ _Float16 As[BM * APITCH];
    __shared__ int   s_tok[BM];
    __shared__ float s_w[BM];

    int t = blockIdx.x;
    int e = tile_e[t];
    if (e < 0) return;                              // uniform: padded tail tile

    int tid = threadIdx.x;
    if (tid < BM) {
        s_tok[tid] = slot_tok[t * BM + tid];
        s_w[tid]   = slot_w[t * BM + tid];
    }
    __syncthreads();

    int wave = tid >> 5, lane = tid & 31;
    int half = lane >> 4, nl = lane & 15;
    int h0 = blockIdx.y * BN;
    const _Float16* W = w16 + (long long)e * HDIM * DDIM;

    // A staging: thread -> (row, 16-float col segment)
    int ar = tid >> 3, ac = (tid & 7) * 16;
    int atok = s_tok[ar];
    const float* arow = (atok >= 0) ? (x + (long long)atok * DDIM + ac)
                                    : (const float*)nullptr;
    _Float16* asd = As + ar * APITCH + ac;

    // per-lane fragment pointers (ISA 16-bit A 16x32 / B 32x16 wave32 layouts)
    const _Float16* a0p = As + nl * APITCH + half * 8;          // m-tile 0
    const _Float16* a1p = As + (16 + nl) * APITCH + half * 8;   // m-tile 1
    const _Float16* bp  = W + (long long)(h0 + wave * 16 + nl) * DDIM + half * 8;

    v8f c0 = {}, c1 = {};
    for (int k0 = 0; k0 < DDIM; k0 += BK) {
        // stage A tile (f32 -> f16)
        if (arow) {
            const float4* a4 = (const float4*)(arow + k0);
#pragma unroll
            for (int q = 0; q < 4; q++) {
                float4 v = a4[q];
                asd[q * 4 + 0] = (_Float16)v.x;
                asd[q * 4 + 1] = (_Float16)v.y;
                asd[q * 4 + 2] = (_Float16)v.z;
                asd[q * 4 + 3] = (_Float16)v.w;
            }
        } else {
#pragma unroll
            for (int j = 0; j < 16; j++) asd[j] = (_Float16)0.0f;
        }
        __syncthreads();

#pragma unroll
        for (int kb = 0; kb < BK / 32; kb++) {
            int ko = kb * 32;
            v8h blo = *(const v8h*)(bp + k0 + ko);
            v8h bhi = *(const v8h*)(bp + k0 + ko + 16);
            v16h a0, a1, bf;
#pragma unroll
            for (int j = 0; j < 8; j++) {
                bf[j]     = blo[j];
                bf[j + 8] = bhi[j];
                a0[j]     = a0p[ko + j];
                a0[j + 8] = a0p[ko + 16 + j];
                a1[j]     = a1p[ko + j];
                a1[j + 8] = a1p[ko + 16 + j];
            }
            c0 = __builtin_amdgcn_wmma_f32_16x16x32_f16(false, a0, false, bf,
                                                        (short)0, c0, false, false);
            c1 = __builtin_amdgcn_wmma_f32_16x16x32_f16(false, a1, false, bf,
                                                        (short)0, c1, false, false);
        }
        __syncthreads();
    }

    int hcol = h0 + wave * 16 + nl;
    float bias = eb[(long long)e * HDIM + hcol];
#pragma unroll
    for (int r = 0; r < 8; r++) {
        int m = r + half * 8;
        int tok = s_tok[m];
        if (tok >= 0)
            unsafeAtomicAdd(&out[(long long)tok * HDIM + hcol],
                            s_w[m] * (c0[r] + bias));
        int m1 = m + 16;
        tok = s_tok[m1];
        if (tok >= 0)
            unsafeAtomicAdd(&out[(long long)tok * HDIM + hcol],
                            s_w[m1] * (c1[r] + bias));
    }
}

extern "C" void kernel_launch(void* const* d_in, const int* in_sizes, int n_in,
                              void* d_out, int out_size, void* d_ws, size_t ws_size,
                              hipStream_t stream) {
    const float* x  = (const float*)d_in[0];
    const float* gw = (const float*)d_in[1];
    const float* gb = (const float*)d_in[2];
    const float* ew = (const float*)d_in[3];
    const float* eb = (const float*)d_in[4];
    float* out = (float*)d_out;
    int*   wsi = (int*)d_ws;
    float* wsf = (float*)d_ws;
    _Float16* w16 = (_Float16*)((char*)d_ws + OFF_W16_BYTES);

    k_init<<<2048, 256, 0, stream>>>(out, wsi, wsf);
    k_cvt<<<8192, 256, 0, stream>>>(ew, w16);
    k_gate<<<B_TOK / 8, 256, 0, stream>>>(x, gw, gb,
                                          wsi + OFF_ROUTE_E,
                                          wsf + OFF_ROUTE_P,
                                          wsi + OFF_COUNTS);
    k_scan<<<1, 32, 0, stream>>>(wsi);
    k_scatter<<<B_TOK / 256, 256, 0, stream>>>(wsi, wsf);
    k_gemm<<<dim3(MAXTILES, HDIM / BN), 256, 0, stream>>>(
        x, w16, eb,
        wsi + OFF_SLOT_TOK, wsf + OFF_SLOT_W, wsi + OFF_TILE_E, out);
}
